// MultiheadLshAttention_21869973471241
// MI455X (gfx1250) — matrix operand order repaired and checked by reading, not compiled, not run on hardware
//
#include <hip/hip_runtime.h>

typedef __bf16 bf16;
typedef unsigned int u32;
typedef unsigned long long u64;
typedef __attribute__((ext_vector_type(16))) __bf16 v16bf;
typedef __attribute__((ext_vector_type(8)))  __bf16 v8bf;
typedef __attribute__((ext_vector_type(8)))  float  v8f;
typedef __attribute__((ext_vector_type(4)))  u32    v4u;
typedef __attribute__((ext_vector_type(8)))  u32    v8u;

#define S_ 4096
#define B_ 2
#define E_ 1024
#define H_ 8
#define R_ 2
#define D_ 128
#define C_ 64
#define NC_ 64
#define SCALE_ 0.088388347648318447f   /* 128^-0.5 */
#define NEG_ (-1.0e9f)
#define NEGSELF_ (-1.0e5f)

// ---------------------------------------------------------------------------
// helpers
// ---------------------------------------------------------------------------
static __device__ __forceinline__ float wave_allreduce_add(float v) {
#pragma unroll
  for (int off = 16; off > 0; off >>= 1) v += __shfl_xor(v, off, 32);
  return v;
}

// Build a 16x32 bf16 A-fragment for one lane from a K-contiguous row.
// Layout: element e -> K = hi*16 + e (contiguous, matches B fragments)
static __device__ __forceinline__ v16bf load_a_frag(const bf16* rowp, int hi) {
  v8bf r0 = *(const v8bf*)(rowp + hi * 16);
  v8bf r1 = *(const v8bf*)(rowp + hi * 16 + 8);
  v16bf a;
#pragma unroll
  for (int i = 0; i < 8; i++) { a[i] = r0[i]; a[8 + i] = r1[i]; }
  return a;
}

// Issue a TDM 2D tile load (Global -> LDS), bf16 elements (data_size=2B).
// D# per CDNA5 ISA 8.3-8.6; groups placed in SGPRs, VADDR2/3 = zero groups.
static __device__ __forceinline__ void tdm_load_tile_2d(u32 lds_byte_addr,
                                                        const void* gptr,
                                                        u32 tensor_d0, u32 tensor_d1,
                                                        u32 tile_d0, u32 tile_d1,
                                                        u32 stride_d0_elems) {
  u64 ga = (u64)(uintptr_t)gptr;
  v4u g0;
  g0[0] = 1u;                                          // count=1 (valid user D#)
  g0[1] = lds_byte_addr;                               // lds_addr
  g0[2] = (u32)ga;                                     // global_addr[31:0]
  g0[3] = (u32)((ga >> 32) & 0x1FFFFFFu) | (2u << 30); // global_addr[56:32] | type=2
  v8u g1;
  g1[0] = (1u << 16);                                  // mask=0, data_size=1 (2B)
  g1[1] = (tensor_d0 & 0xFFFFu) << 16;                 // tensor_dim0[15:0] @ bits63:48
  g1[2] = (tensor_d0 >> 16) | ((tensor_d1 & 0xFFFFu) << 16);
  g1[3] = (tensor_d1 >> 16) | ((tile_d0 & 0xFFFFu) << 16);   // tile_dim0 @ 127:112
  g1[4] = tile_d1 & 0xFFFFu;                           // tile_dim1, tile_dim2=0
  g1[5] = stride_d0_elems;                             // tensor_dim0_stride[31:0]
  g1[6] = 0;                                           // stride0 hi, stride1 lo
  g1[7] = 0;
  v4u gz = (v4u)(0u);
  asm volatile("tensor_load_to_lds %0, %1, %2, %3"
               :: "s"(g0), "s"(g1), "s"(gz), "s"(gz) : "memory");
}

// Per-lane async global->LDS 16B copy (ASYNCcnt-tracked).
static __device__ __forceinline__ void async_copy_b128(u32 lds_byte_addr,
                                                       const void* gptr) {
  asm volatile("global_load_async_to_lds_b128 %0, %1, off"
               :: "v"(lds_byte_addr), "v"(gptr) : "memory");
}
static __device__ __forceinline__ void wait_asynccnt0() {
  asm volatile("s_wait_asynccnt 0x0" ::: "memory");
}

// ---------------------------------------------------------------------------
// f32 -> bf16 convert
// ---------------------------------------------------------------------------
__global__ __launch_bounds__(256) void f32_to_bf16_kernel(const float* __restrict__ src,
                                                          bf16* __restrict__ dst, int n) {
  int i = blockIdx.x * 256 + threadIdx.x;
  if (i < n) dst[i] = (bf16)src[i];
}

// ---------------------------------------------------------------------------
// NT GEMM: C[M,N] = A[M,K] * B[N,K]^T + bias[N]   (A,B bf16, acc f32)
// 256 threads (8 waves), block tile 128x128, wave tile 64x32.
// A/B 128x32 tiles staged to LDS by the Tensor Data Mover, double-buffered.
// ---------------------------------------------------------------------------
template <bool OUT_BF16>
__global__ __launch_bounds__(256) void gemm_nt_bf16(const bf16* __restrict__ A,
                                                    const bf16* __restrict__ Bm,
                                                    const float* __restrict__ bias,
                                                    void* __restrict__ Cout,
                                                    int M, int N, int K) {
  __shared__ bf16 Abuf[2][128 * 32];
  __shared__ bf16 Bbuf[2][128 * 32];

  const int lane = threadIdx.x & 31;
  const int wid  = threadIdx.x >> 5;
  const int wm = wid & 1, wn = wid >> 1;                 // 2 x 4 wave grid
  const int m0g = blockIdx.y * 128;
  const int n0g = blockIdx.x * 128;
  const int hi = lane >> 4, lm = lane & 15;

  v8f z = {};
  v8f acc[4][2];
#pragma unroll
  for (int t = 0; t < 4; t++)
#pragma unroll
    for (int j = 0; j < 2; j++) acc[t][j] = z;

  // prologue: stage k0=0 tiles into buffer 0
  if (wid == 0) {
    tdm_load_tile_2d((u32)(uintptr_t)&Abuf[0][0], A + (size_t)m0g * K,
                     (u32)K, (u32)M, 32, 128, (u32)K);
    tdm_load_tile_2d((u32)(uintptr_t)&Bbuf[0][0], Bm + (size_t)n0g * K,
                     (u32)K, (u32)N, 32, 128, (u32)K);
  }

  int buf = 0;
  for (int k0 = 0; k0 < K; k0 += 32) {
    if (wid == 0) {
      if (k0 + 32 < K) {
        tdm_load_tile_2d((u32)(uintptr_t)&Abuf[buf ^ 1][0],
                         A + (size_t)m0g * K + (k0 + 32),
                         (u32)K, (u32)M, 32, 128, (u32)K);
        tdm_load_tile_2d((u32)(uintptr_t)&Bbuf[buf ^ 1][0],
                         Bm + (size_t)n0g * K + (k0 + 32),
                         (u32)K, (u32)N, 32, 128, (u32)K);
        __builtin_amdgcn_s_wait_tensorcnt(2);   // current pair landed
      } else {
        __builtin_amdgcn_s_wait_tensorcnt(0);
      }
    }
    __syncthreads();

    v16bf a[4];
#pragma unroll
    for (int t = 0; t < 4; t++)
      a[t] = load_a_frag(&Abuf[buf][(wm * 64 + t * 16 + lm) * 32], hi);
    v16bf bb[2];
#pragma unroll
    for (int j = 0; j < 2; j++)
      bb[j] = *(const v16bf*)&Bbuf[buf][(wn * 32 + j * 16 + lm) * 32 + hi * 16];
#pragma unroll
    for (int t = 0; t < 4; t++)
#pragma unroll
      for (int j = 0; j < 2; j++)
        acc[t][j] = __builtin_amdgcn_wmma_f32_16x16x32_bf16(
            false, a[t], false, bb[j], (short)0, acc[t][j], false, false);

    __syncthreads();  // all waves done with buf before TDM overwrites it
    buf ^= 1;
  }

#pragma unroll
  for (int t = 0; t < 4; t++)
#pragma unroll
    for (int j = 0; j < 2; j++)
#pragma unroll
      for (int g = 0; g < 8; g++) {
        int row = m0g + wm * 64 + t * 16 + g + hi * 8;
        int col = n0g + wn * 32 + j * 16 + lm;
        float v = acc[t][j][g] + bias[col];
        if constexpr (OUT_BF16)
          ((bf16*)Cout)[(size_t)row * N + col] = (bf16)v;
        else
          ((float*)Cout)[(size_t)row * N + col] = v;
      }
}

// ---------------------------------------------------------------------------
// Per-token (s,b): norm over full E=1024, write qb = q*SCALE, kb = q/||q||
// layouts: q_raw [S,B,E] f32 ; qb,kb [B,H,S,D] bf16.  one wave per token.
// ---------------------------------------------------------------------------
__global__ __launch_bounds__(256) void norm_project_kernel(const float* __restrict__ q_raw,
                                                           bf16* __restrict__ qb,
                                                           bf16* __restrict__ kb) {
  const int lane = threadIdx.x & 31;
  const int wid  = threadIdx.x >> 5;
  const int t = blockIdx.x * 8 + wid;          // row index = s*B + b
  const int s = t / B_, b = t % B_;
  const float* row = q_raw + (size_t)t * E_;

  float vals[32];
  float ss = 0.f;
#pragma unroll
  for (int i = 0; i < 32; i++) {
    float v = row[lane + 32 * i];
    vals[i] = v;
    ss += v * v;
  }
  ss = wave_allreduce_add(ss);
  float rn = rsqrtf(ss);

#pragma unroll
  for (int i = 0; i < 32; i++) {
    int e = lane + 32 * i;
    int h = e >> 7, d = e & 127;
    size_t o = ((size_t)(b * H_ + h) * S_ + s) * D_ + d;
    qb[o] = (bf16)(vals[i] * SCALE_);
    kb[o] = (bf16)(vals[i] * rn);
  }
}

// ---------------------------------------------------------------------------
// LSH hashing: one wave per (b,h,s); hashes[b,r,h,s] = argmax([lin,-lin])
// hash_w: [R,H,D,8] f32
// ---------------------------------------------------------------------------
__global__ __launch_bounds__(256) void lsh_hash_kernel(const bf16* __restrict__ kb,
                                                       const float* __restrict__ hw,
                                                       int* __restrict__ hashes) {
  const int lane = threadIdx.x & 31;
  const int wid  = threadIdx.x >> 5;
  const int t = blockIdx.x * 8 + wid;          // (b*H+h)*S + s
  const int s = t & (S_ - 1);
  const int bh = t >> 12;
  const int b = bh >> 3, h = bh & 7;

  const bf16* kr = kb + (size_t)t * D_;
  float kd[4];
#pragma unroll
  for (int j = 0; j < 4; j++) kd[j] = (float)kr[lane * 4 + j];

#pragma unroll
  for (int r = 0; r < R_; r++) {
    float lin[8];
#pragma unroll
    for (int m = 0; m < 8; m++) {
      float p = 0.f;
#pragma unroll
      for (int j = 0; j < 4; j++) {
        int d = lane * 4 + j;
        p += kd[j] * hw[((size_t)(r * H_ + h) * D_ + d) * 8 + m];
      }
      lin[m] = wave_allreduce_add(p);
    }
    if (lane == 0) {
      float best = lin[0];
      int bi = 0;
#pragma unroll
      for (int m = 1; m < 8; m++)
        if (lin[m] > best) { best = lin[m]; bi = m; }
#pragma unroll
      for (int m = 0; m < 8; m++) {
        float v = -lin[m];
        if (v > best) { best = v; bi = 8 + m; }
      }
      hashes[((size_t)(b * R_ + r) * H_ + h) * S_ + s] = bi;
    }
  }
}

// ---------------------------------------------------------------------------
// Stable counting sort per (b,r,h): 16 buckets, 4096 keys, 256 threads.
// ---------------------------------------------------------------------------
__global__ __launch_bounds__(256) void lsh_sort_kernel(const int* __restrict__ hashes,
                                                       int* __restrict__ sort_idx,
                                                       int* __restrict__ inv,
                                                       int* __restrict__ h_sorted) {
  __shared__ int cnt[16 * 256];
  __shared__ int tsum[256];
  const int t = threadIdx.x;
  const size_t base = (size_t)blockIdx.x * S_;

  int keys[16];
  int local[16];
#pragma unroll
  for (int k = 0; k < 16; k++) local[k] = 0;
#pragma unroll
  for (int i = 0; i < 16; i++) {
    int k = hashes[base + t * 16 + i];
    keys[i] = k;
    local[k]++;
  }
#pragma unroll
  for (int k = 0; k < 16; k++) cnt[k * 256 + t] = local[k];
  __syncthreads();

  int run = 0;
#pragma unroll
  for (int i = 0; i < 16; i++) {
    int v = cnt[t * 16 + i];
    cnt[t * 16 + i] = run;
    run += v;
  }
  tsum[t] = run;
  __syncthreads();
  int incl = run;
  for (int off = 1; off < 256; off <<= 1) {
    int add = (t >= off) ? tsum[t - off] : 0;
    __syncthreads();
    incl += add;
    tsum[t] = incl;
    __syncthreads();
  }
  int excl = incl - run;
#pragma unroll
  for (int i = 0; i < 16; i++) cnt[t * 16 + i] += excl;
  __syncthreads();

  int start[16];
#pragma unroll
  for (int k = 0; k < 16; k++) start[k] = cnt[k * 256 + t];
#pragma unroll
  for (int i = 0; i < 16; i++) {
    int spos = t * 16 + i;
    int k = keys[i];
    int dst = start[k]++;
    sort_idx[base + dst] = spos;
    h_sorted[base + dst] = k;
    inv[base + spos] = dst;
  }
}

// ---------------------------------------------------------------------------
// Chunked LSH attention. One block per (b,r,h,n): 256 threads (8 waves).
//   K window staged to LDS via global_load_async_to_lds_b128 (ASYNCcnt)
//   gemm1: logits[64,192] = Qc(gathered) * Kw^T        (WMMA bf16)
//   masks + row softmax/LSE; V staged transposed to LDS (overlapping KW region)
//   gemm2: O[64,128] = P * Vw                          (WMMA bf16)
// LDS: Lg f32[64][196] | P bf16[64][192] | KW[192][144] / VT[128][208] | idx
// ---------------------------------------------------------------------------
#define SMEM_L  (64 * 196 * 4)
#define SMEM_P  (64 * 192 * 2)
#define SMEM_KV (192 * 144 * 2)   /* union of KW (55296B) and VT (53248B) */
#define SMEM_I  ((64 + 64 + 192 + 192) * 4)
#define SMEM_TOTAL (SMEM_L + SMEM_P + SMEM_KV + SMEM_I)

__global__ __launch_bounds__(256) void lsh_attention_kernel(
    const bf16* __restrict__ qb, const bf16* __restrict__ kb,
    const bf16* __restrict__ vb, const int* __restrict__ sort_idx,
    const int* __restrict__ h_sorted, float* __restrict__ l_s,
    float* __restrict__ o_srt) {
  extern __shared__ char smem[];
  float* Lg = (float*)smem;                          // [64][196]
  bf16*  P  = (bf16*)(smem + SMEM_L);                // [64][192]
  bf16*  KW = (bf16*)(smem + SMEM_L + SMEM_P);       // [192][144] (w-major)
  bf16*  VT = (bf16*)(smem + SMEM_L + SMEM_P);       // [128][208] (d-major), reuses KW
  int*   HQ = (int*)(smem + SMEM_L + SMEM_P + SMEM_KV);
  int*   IQ = HQ + 64;
  int*   HW = IQ + 64;
  int*   IW = HW + 192;

  const int blk = blockIdx.x;           // ((b*R+r)*H + h)*NC + n
  const int n   = blk & (NC_ - 1);
  const int brh = blk >> 6;             // (b*R+r)*H + h
  const int h   = brh & (H_ - 1);
  const int b   = brh >> 4;             // R_*H_ == 16
  const int bh  = b * H_ + h;
  const size_t base = (size_t)brh * S_;
  const int tid = threadIdx.x;

  if (tid < 64) {
    int sp = n * 64 + tid;
    IQ[tid] = sort_idx[base + sp];
    HQ[tid] = h_sorted[base + sp];
  }
  if (tid < 192) {
    int w = tid;
    int c = (n + (w >> 6) - 1 + NC_) & (NC_ - 1);
    int sp = c * 64 + (w & 63);
    IW[w] = sort_idx[base + sp];
    HW[w] = h_sorted[base + sp];
  }
  __syncthreads();

  // async-stage K window rows into LDS (12 x 16B per thread)
  const bf16* kbase = kb + (size_t)bh * S_ * D_;
  const u32 kw_base = (u32)(uintptr_t)KW;
  for (int e = tid; e < 192 * 16; e += 256) {
    int w = e >> 4, seg = e & 15;
    async_copy_b128(kw_base + (u32)(w * 144 + seg * 8) * 2,
                    kbase + (size_t)IW[w] * D_ + seg * 8);
  }
  wait_asynccnt0();
  __syncthreads();

  const int lane = tid & 31, wid = tid >> 5;
  const int hi = lane >> 4, lm = lane & 15;
  const int mt = wid & 3;      // row tile 16*mt
  const int cg = wid >> 2;     // col group (0/1)
  v8f z = {};

  // ---- gemm1: logits = Q * Kw^T, contraction over D=128 ----
  {
    v8f c1[6];
#pragma unroll
    for (int j = 0; j < 6; j++) c1[j] = z;

    const bf16* qbase = qb + (size_t)bh * S_ * D_;
    const bf16* ap0 = qbase + (size_t)IQ[mt * 16 + lm] * D_;
    for (int k0 = 0; k0 < 128; k0 += 32) {
      v16bf a = load_a_frag(ap0 + k0, hi);
#pragma unroll
      for (int j = 0; j < 6; j++) {
        int wcol = cg * 96 + j * 16 + lm;
        v16bf bb = *(const v16bf*)(KW + wcol * 144 + k0 + hi * 16);
        c1[j] = __builtin_amdgcn_wmma_f32_16x16x32_bf16(false, a, false, bb,
                                                        (short)0, c1[j], false, false);
      }
    }
#pragma unroll
    for (int j = 0; j < 6; j++)
#pragma unroll
      for (int g = 0; g < 8; g++) {
        int row = mt * 16 + g + hi * 8;
        int col = cg * 96 + j * 16 + lm;
        float v = c1[j][g];
        if (HQ[row] != HW[col]) v = NEG_;
        if (IQ[row] == IW[col]) v = NEGSELF_;
        Lg[row * 196 + col] = v;
      }
  }
  __syncthreads();   // gemm1 done: KW region free, Lg visible

  // ---- stage V transposed (over KW region) + softmax/LSE ----
  for (int e = tid; e < 192 * 128; e += 256) {
    int w = e >> 7, d = e & 127;
    int tok = IW[w];
    VT[d * 208 + w] = vb[((size_t)tok * B_ + b) * E_ + h * D_ + d];
  }
  if (tid < 64) {
    int row = tid;
    float mx = -3.0e38f;
    for (int c = 0; c < 192; c++) mx = fmaxf(mx, Lg[row * 196 + c]);
    float ssum = 0.f;
    for (int c = 0; c < 192; c++) {
      float e = __expf(Lg[row * 196 + c] - mx);
      ssum += e;
      Lg[row * 196 + c] = e;
    }
    float is = 1.f / ssum;
    for (int c = 0; c < 192; c++) P[row * 192 + c] = (bf16)(Lg[row * 196 + c] * is);
    l_s[base + n * 64 + row] = mx + __logf(ssum);
  }
  __syncthreads();

  // ---- gemm2: O = P * Vw, contraction over window=192 ----
  {
    v8f c2[4];
#pragma unroll
    for (int j = 0; j < 4; j++) c2[j] = z;

    for (int k0 = 0; k0 < 192; k0 += 32) {
      v16bf a = load_a_frag(P + (mt * 16 + lm) * 192 + k0, hi);
#pragma unroll
      for (int j = 0; j < 4; j++) {
        int d = cg * 64 + j * 16 + lm;
        v16bf bb = *(const v16bf*)(VT + d * 208 + k0 + hi * 16);
        c2[j] = __builtin_amdgcn_wmma_f32_16x16x32_bf16(false, a, false, bb,
                                                        (short)0, c2[j], false, false);
      }
    }
#pragma unroll
    for (int j = 0; j < 4; j++)
#pragma unroll
      for (int g = 0; g < 8; g++) {
        int row = mt * 16 + g + hi * 8;
        int d = cg * 64 + j * 16 + lm;
        o_srt[(base + n * 64 + row) * D_ + d] = c2[j][g];
      }
  }
}

// ---------------------------------------------------------------------------
// Combine rounds: softmax over per-round LSE; scatter back via inverse perm.
// ---------------------------------------------------------------------------
__global__ __launch_bounds__(256) void combine_rounds_kernel(const float* __restrict__ o_srt,
                                                             const float* __restrict__ l_s,
                                                             const int* __restrict__ inv,
                                                             bf16* __restrict__ ctx) {
  size_t t = (size_t)blockIdx.x * 256 + threadIdx.x;   // over B*H*S*D
  int d = (int)(t & 127);
  size_t r1 = t >> 7;
  int s = (int)(r1 & (S_ - 1));
  size_t bhs = r1 >> 12;
  int h = (int)(bhs & 7);
  int b = (int)(bhs >> 3);

  size_t i0 = ((size_t)(b * R_ + 0) * H_ + h) * S_;
  size_t i1 = ((size_t)(b * R_ + 1) * H_ + h) * S_;
  int p0 = inv[i0 + s], p1 = inv[i1 + s];
  float l0 = l_s[i0 + p0], l1 = l_s[i1 + p1];
  float m = fmaxf(l0, l1);
  float e0 = __expf(l0 - m), e1 = __expf(l1 - m);
  float isum = 1.f / (e0 + e1);
  float o0 = o_srt[(i0 + p0) * D_ + d];
  float o1 = o_srt[(i1 + p1) * D_ + d];
  ctx[((size_t)s * B_ + b) * E_ + h * D_ + d] = (bf16)((e0 * o0 + e1 * o1) * isum);
}

// ---------------------------------------------------------------------------
// host launcher
// ---------------------------------------------------------------------------
extern "C" void kernel_launch(void* const* d_in, const int* in_sizes, int n_in,
                              void* d_out, int out_size, void* d_ws, size_t ws_size,
                              hipStream_t stream) {
  (void)in_sizes; (void)n_in; (void)out_size; (void)ws_size;
  const float* x      = (const float*)d_in[0];
  const float* wq     = (const float*)d_in[1];
  const float* bq     = (const float*)d_in[2];
  const float* wv     = (const float*)d_in[3];
  const float* bv     = (const float*)d_in[4];
  const float* wo     = (const float*)d_in[5];
  const float* bo     = (const float*)d_in[6];
  const float* hash_w = (const float*)d_in[7];

  char* ws = (char*)d_ws;
  size_t off = 0;
  auto carve = [&](size_t bytes) -> char* {
    char* p = ws + off;
    off += (bytes + 255) & ~(size_t)255;
    return p;
  };
  const size_t MN = (size_t)S_ * B_ * E_;     // 8,388,608
  bf16*  xb     = (bf16*)carve(MN * 2);
  bf16*  wqb    = (bf16*)carve((size_t)E_ * E_ * 2);
  bf16*  wvb    = (bf16*)carve((size_t)E_ * E_ * 2);
  bf16*  wob    = (bf16*)carve((size_t)E_ * E_ * 2);
  float* q_raw  = (float*)carve(MN * 4);
  bf16*  vb     = (bf16*)carve(MN * 2);
  bf16*  qb     = (bf16*)carve(MN * 2);
  bf16*  kb     = (bf16*)carve(MN * 2);
  int*   hashes = (int*)carve((size_t)B_ * R_ * H_ * S_ * 4);
  int*   sidx   = (int*)carve((size_t)B_ * R_ * H_ * S_ * 4);
  int*   invp   = (int*)carve((size_t)B_ * R_ * H_ * S_ * 4);
  int*   hsort  = (int*)carve((size_t)B_ * R_ * H_ * S_ * 4);
  float* l_s    = (float*)carve((size_t)B_ * R_ * H_ * S_ * 4);
  float* o_srt  = (float*)carve((size_t)B_ * R_ * H_ * S_ * D_ * 4);
  bf16*  ctxb   = (bf16*)carve(MN * 2);

  // 1) bf16 conversions
  f32_to_bf16_kernel<<<(int)(MN / 256), 256, 0, stream>>>(x, xb, (int)MN);
  f32_to_bf16_kernel<<<E_ * E_ / 256, 256, 0, stream>>>(wq, wqb, E_ * E_);
  f32_to_bf16_kernel<<<E_ * E_ / 256, 256, 0, stream>>>(wv, wvb, E_ * E_);
  f32_to_bf16_kernel<<<E_ * E_ / 256, 256, 0, stream>>>(wo, wob, E_ * E_);

  // 2) projections: q_raw (f32) and v (bf16)   [TDM-staged WMMA GEMM]
  dim3 ggrid(E_ / 128, (S_ * B_) / 128);   // (8, 64)
  gemm_nt_bf16<false><<<ggrid, 256, 0, stream>>>(xb, wqb, bq, (void*)q_raw,
                                                 S_ * B_, E_, E_);
  gemm_nt_bf16<true><<<ggrid, 256, 0, stream>>>(xb, wvb, bv, (void*)vb,
                                                S_ * B_, E_, E_);

  // 3) full-row normalization, head split, scaling
  norm_project_kernel<<<(S_ * B_) / 8, 256, 0, stream>>>(q_raw, qb, kb);

  // 4) LSH hashing
  lsh_hash_kernel<<<(B_ * H_ * S_) / 8, 256, 0, stream>>>(kb, hash_w, hashes);

  // 5) stable counting sort per (b,r,h)
  lsh_sort_kernel<<<B_ * R_ * H_, 256, 0, stream>>>(hashes, sidx, invp, hsort);

  // 6) chunked attention (WMMA + async LDS staging)
  lsh_attention_kernel<<<B_ * R_ * H_ * NC_, 256, SMEM_TOTAL, stream>>>(
      qb, kb, vb, sidx, hsort, l_s, o_srt);

  // 7) combine rounds + un-permute -> ctx (bf16)
  combine_rounds_kernel<<<(int)((size_t)B_ * H_ * S_ * D_ / 256), 256, 0, stream>>>(
      o_srt, l_s, invp, ctxb);

  // 8) output projection -> d_out (f32)
  gemm_nt_bf16<false><<<ggrid, 256, 0, stream>>>(ctxb, wob, bo, d_out,
                                                 S_ * B_, E_, E_);
}
